// DCRNNModel_47974784696676
// MI455X (gfx1250) — compile-verified, matrix operand order checked by reading, not run on hardware
//
#include <hip/hip_runtime.h>

// ---------------------------------------------------------------------------
// DCRNN forward on gfx1250: bf16 WMMA (v_wmma_f32_16x16x32_bf16) GEMMs for the
// diffusion (S @ X) and weight matmuls, f32 state/epilogues.
// Diffusion GEMM uses async global->LDS copies (ASYNCcnt) with LDS double
// buffering when the toolchain exposes the gfx1250 async builtins.
// ---------------------------------------------------------------------------

typedef __attribute__((ext_vector_type(16))) __bf16 v16bf;
typedef __attribute__((ext_vector_type(8)))  float  v8f;
typedef __attribute__((ext_vector_type(4)))  int    iv4;

static constexpr int kN   = 1024;     // nodes
static constexpr int kB   = 32;       // batch
static constexpr int kU   = 64;       // rnn units
static constexpr int kM   = kN * kB;  // 32768 rows (node-major, batch-minor)
static constexpr int kSeq = 12;
static constexpr int kHor = 12;
static constexpr int LDAP = 40;       // LDS tile pitch in bf16 elems (80B: 16B-aligned, low-conflict)

#if defined(__has_builtin)
#  if __has_builtin(__builtin_amdgcn_global_load_async_to_lds_b128) && \
      __has_builtin(__builtin_amdgcn_s_wait_asynccnt)
#    define DC_ASYNC 1
#  endif
#endif
#ifndef DC_ASYNC
#  define DC_ASYNC 0
#endif

#if DC_ASYNC
typedef __attribute__((address_space(1))) iv4* gas_iv4p;  // global int4*
typedef __attribute__((address_space(3))) iv4* las_iv4p;  // LDS int4*
__device__ __forceinline__ void async_cp16(const unsigned short* g, unsigned short* l) {
  // LDS[l] = MEM[g], 16 bytes per lane, tracked by ASYNCcnt
  __builtin_amdgcn_global_load_async_to_lds_b128((gas_iv4p)(void*)g, (las_iv4p)(void*)l, 0, 0);
}
__device__ __forceinline__ void async_wait0() { __builtin_amdgcn_s_wait_asynccnt(0); }
#endif

__device__ __forceinline__ unsigned short f2bf(float f) {
  union { float f; unsigned u; } v; v.f = f;
  unsigned r = v.u + 0x7FFFu + ((v.u >> 16) & 1u);   // RNE
  return (unsigned short)(r >> 16);
}
__device__ __forceinline__ float bf2f(unsigned short h) {
  union { unsigned u; float f; } v; v.u = ((unsigned)h) << 16; return v.f;
}

// Load one 16x32 bf16 WMMA fragment (A or B^T) from an LDS tile stored
// row-major with pitch LDAP. Per ISA layout: lane<16 holds K{0..7,16..23},
// lane>=16 holds K{8..15,24..31} of row (lane&15). Two 16B ds loads.
__device__ __forceinline__ v16bf frag_ld(const unsigned short* base, int row0, int lane) {
  const int l  = lane & 15;
  const int hi = (lane >> 4) & 1;
  const unsigned short* p = base + (row0 + l) * LDAP + hi * 8;
  union { uint4 q[2]; v16bf v; } u;
  u.q[0] = *(const uint4*)(p);
  u.q[1] = *(const uint4*)(p + 16);
  return u.v;
}

__device__ __forceinline__ v8f wmma_bf16(v16bf a, v16bf b, v8f c) {
  return __builtin_amdgcn_wmma_f32_16x16x32_bf16(false, a, false, b, (short)0, c,
                                                 false, false);
}

// ---------------------------------------------------------------------------
// Diffusion GEMM: Y(1024 x ldx) = S(1024x1024) * X(1024 x ldx)       [cheb==0]
//                 Y = 2*S*X - Xsub                                    [cheb==1]
// ldx is padded to a multiple of 128 (pad columns are zero), so there are no
// bounds checks anywhere. bf16 in, f32 accumulate, bf16 out.
// BM=BN=128, BK=32, 8 waves * (2x4) WMMA tiles, LDS double buffered.
// ---------------------------------------------------------------------------
__global__ __launch_bounds__(256) void gemm_diffuse(
    const unsigned short* __restrict__ S,
    const unsigned short* __restrict__ X,
    const unsigned short* __restrict__ Xsub,
    unsigned short* __restrict__ Y,
    int ldx, int cheb)
{
  __shared__ __align__(16) unsigned short As[2][128 * LDAP];
  __shared__ __align__(16) unsigned short Bs[2][128 * LDAP];
  const int tid  = threadIdx.x;
  const int lane = tid & 31;
  const int wave = tid >> 5;
  const int wm = wave >> 1;          // 0..3 : 32 rows each
  const int wn = wave & 1;           // 0..1 : 64 cols each
  const int mBase = blockIdx.y * 128;
  const int nBase = blockIdx.x * 128;

  v8f acc[2][4];
#pragma unroll
  for (int i = 0; i < 2; ++i)
#pragma unroll
    for (int j = 0; j < 4; ++j)
#pragma unroll
      for (int e = 0; e < 8; ++e) acc[i][j][e] = 0.f;

  const int arow = tid >> 1, ahalf = tid & 1;  // A: 128 rows x 2 half-rows
  const int bk = tid >> 3, bng = tid & 7;      // B: 32 k-rows x 8 col-groups
  const int aoff = arow * LDAP + ahalf * 16;   // LDS A slot for this thread
  const unsigned short* gA = S + (size_t)(mBase + arow) * kN + ahalf * 16;
  const unsigned short* gB = X + (size_t)bk * ldx + nBase + bng * 16;

  // ---- prologue: stage tile 0 ----
#if DC_ASYNC
  async_cp16(gA,     &As[0][aoff]);
  async_cp16(gA + 8, &As[0][aoff + 8]);
#else
  *(uint4*)&As[0][aoff]     = *(const uint4*)gA;
  *(uint4*)&As[0][aoff + 8] = *(const uint4*)(gA + 8);
#endif
  {
    unsigned short t0[16];
    *(uint4*)&t0[0] = *(const uint4*)gB;
    *(uint4*)&t0[8] = *(const uint4*)(gB + 8);
#if DC_ASYNC
    async_wait0();
#endif
#pragma unroll
    for (int j = 0; j < 16; ++j) Bs[0][(bng * 16 + j) * LDAP + bk] = t0[j];
  }
  __syncthreads();

  constexpr int kTiles = kN / 32;    // 32
  for (int it = 0; it < kTiles; ++it) {
    const int cb = it & 1, nb = cb ^ 1;
    unsigned short tmpB[16];
    const bool more = (it + 1) < kTiles;
    if (more) {
      // overlap next tile's loads with this tile's WMMAs
      const unsigned short* ga = gA + (size_t)(it + 1) * 32;
#if DC_ASYNC
      async_cp16(ga,     &As[nb][aoff]);
      async_cp16(ga + 8, &As[nb][aoff + 8]);
#else
      *(uint4*)&As[nb][aoff]     = *(const uint4*)ga;
      *(uint4*)&As[nb][aoff + 8] = *(const uint4*)(ga + 8);
#endif
      __builtin_prefetch(ga + 64, 0, 3);   // global_prefetch two tiles ahead of ga
      const unsigned short* gb = gB + (size_t)(it + 1) * 32 * ldx;
      *(uint4*)&tmpB[0] = *(const uint4*)gb;
      *(uint4*)&tmpB[8] = *(const uint4*)(gb + 8);
    }

    // compute on current buffers
    v16bf a0 = frag_ld(As[cb], wm * 32,      lane);
    v16bf a1 = frag_ld(As[cb], wm * 32 + 16, lane);
    v16bf b0 = frag_ld(Bs[cb], wn * 64,      lane);
    v16bf b1 = frag_ld(Bs[cb], wn * 64 + 16, lane);
    v16bf b2 = frag_ld(Bs[cb], wn * 64 + 32, lane);
    v16bf b3 = frag_ld(Bs[cb], wn * 64 + 48, lane);

    acc[0][0] = wmma_bf16(a0, b0, acc[0][0]);
    acc[0][1] = wmma_bf16(a0, b1, acc[0][1]);
    acc[0][2] = wmma_bf16(a0, b2, acc[0][2]);
    acc[0][3] = wmma_bf16(a0, b3, acc[0][3]);
    acc[1][0] = wmma_bf16(a1, b0, acc[1][0]);
    acc[1][1] = wmma_bf16(a1, b1, acc[1][1]);
    acc[1][2] = wmma_bf16(a1, b2, acc[1][2]);
    acc[1][3] = wmma_bf16(a1, b3, acc[1][3]);
    __syncthreads();

    if (more) {
#if DC_ASYNC
      async_wait0();                 // next A tile resident in LDS
#endif
#pragma unroll
      for (int j = 0; j < 16; ++j) Bs[nb][(bng * 16 + j) * LDAP + bk] = tmpB[j];
      __syncthreads();
    }
  }

  const int l  = lane & 15;
  const int hi = lane >> 4;
#pragma unroll
  for (int i = 0; i < 2; ++i) {
#pragma unroll
    for (int j = 0; j < 4; ++j) {
      const int col = nBase + wn * 64 + j * 16 + l;
      const int rbase = mBase + wm * 32 + i * 16 + hi * 8;
#pragma unroll
      for (int e = 0; e < 8; ++e) {
        float v = acc[i][j][e];
        const size_t idx = (size_t)(rbase + e) * ldx + col;
        if (cheb) v = 2.f * v - bf2f(Xsub[idx]);
        Y[idx] = f2bf(v);
      }
    }
  }
}

// ---------------------------------------------------------------------------
// Weight GEMM: out(kM x nc) = act( Xk(kM x Kpad, lda=384) * W(Kpad x nc) + b )
// act 0 = sigmoid (gates), act 1 = tanh (candidate). BM=128, BN=64, BK=32.
// ---------------------------------------------------------------------------
__global__ __launch_bounds__(256) void gemm_weight(
    const unsigned short* __restrict__ Xk,
    const unsigned short* __restrict__ W,
    const float* __restrict__ bias,
    float* __restrict__ out,
    int Kpad, int nc, int act)
{
  __shared__ __align__(16) unsigned short As[128 * LDAP];
  __shared__ __align__(16) unsigned short Bs[64 * LDAP];
  const int tid  = threadIdx.x;
  const int lane = tid & 31;
  const int wave = tid >> 5;
  const int wm = wave >> 1;          // 0..3 : 32 rows each
  const int wn = wave & 1;           // 0..1 : 32 cols each
  const int mBase = blockIdx.y * 128;
  const int col0  = blockIdx.x * 64;

  v8f acc[2][2];
#pragma unroll
  for (int i = 0; i < 2; ++i)
#pragma unroll
    for (int j = 0; j < 2; ++j)
#pragma unroll
      for (int e = 0; e < 8; ++e) acc[i][j][e] = 0.f;

  const int arow = tid >> 1, ahalf = tid & 1;
  const int bk = tid >> 3, bng = tid & 7;

  for (int k0 = 0; k0 < Kpad; k0 += 32) {
    { // stage A 128x32
      const unsigned short* g = Xk + (size_t)(mBase + arow) * 384 + k0 + ahalf * 16;
      uint4 q0 = *(const uint4*)g;
      uint4 q1 = *(const uint4*)(g + 8);
      *(uint4*)&As[arow * LDAP + ahalf * 16]     = q0;
      *(uint4*)&As[arow * LDAP + ahalf * 16 + 8] = q1;
      if (k0 + 32 < Kpad) __builtin_prefetch(g + 32, 0, 3);
    }
    { // stage B 32x64 transposed -> Bs[n][k]
      unsigned short tmp[8];
      const unsigned short* g = W + (size_t)(k0 + bk) * nc + col0 + bng * 8;
      *(uint4*)&tmp[0] = *(const uint4*)g;
#pragma unroll
      for (int j = 0; j < 8; ++j) Bs[(bng * 8 + j) * LDAP + bk] = tmp[j];
    }
    __syncthreads();

    v16bf a0 = frag_ld(As, wm * 32,      lane);
    v16bf a1 = frag_ld(As, wm * 32 + 16, lane);
    v16bf b0 = frag_ld(Bs, wn * 32,      lane);
    v16bf b1 = frag_ld(Bs, wn * 32 + 16, lane);

    acc[0][0] = wmma_bf16(a0, b0, acc[0][0]);
    acc[0][1] = wmma_bf16(a0, b1, acc[0][1]);
    acc[1][0] = wmma_bf16(a1, b0, acc[1][0]);
    acc[1][1] = wmma_bf16(a1, b1, acc[1][1]);
    __syncthreads();
  }

  const int l  = lane & 15;
  const int hi = lane >> 4;
#pragma unroll
  for (int i = 0; i < 2; ++i) {
#pragma unroll
    for (int j = 0; j < 2; ++j) {
      const int col = col0 + wn * 32 + j * 16 + l;
      const float bv = bias[col];
      const int rbase = mBase + wm * 32 + i * 16 + hi * 8;
#pragma unroll
      for (int e = 0; e < 8; ++e) {
        float v = acc[i][j][e] + bv;
        v = act ? tanhf(v) : 1.f / (1.f + __expf(-v));
        out[(size_t)(rbase + e) * nc + col] = v;
      }
    }
  }
}

// ---------------------------------------------------------------------------
// Elementwise helpers
// ---------------------------------------------------------------------------
// X0[n, c] over padded width ldx; real columns c = f*kB + b for f<F0, pad = 0.
// modes: 0 enc-L0 (x=(B,N,2)), 1 dec-L0 (x=GO (N*B,1)), 2 L1 (x=lower h).
__global__ void pack_x0(int mode, const float* __restrict__ xsrc,
                        const float* __restrict__ hsrc,
                        unsigned short* __restrict__ X0, int F0, int Din, int ldx)
{
  const int idx = blockIdx.x * 256 + threadIdx.x;
  if (idx >= kN * ldx) return;
  const int c = idx % ldx;
  const int n = idx / ldx;
  float v = 0.f;
  if (c < F0 * kB) {
    const int f = c >> 5;   // kB == 32
    const int b = c & 31;
    if (f < Din) {
      if (mode == 0)      v = xsrc[((size_t)b * kN + n) * 2 + f];
      else if (mode == 1) v = xsrc[(size_t)n * kB + b];
      else                v = xsrc[(size_t)(n * kB + b) * kU + f];
    } else {
      v = hsrc[(size_t)(n * kB + b) * kU + (f - Din)];
    }
  }
  X0[idx] = f2bf(v);
}

// Xk[r, f*3+m] = Xm[n,f,b] (torch stack ordering), zero-padded to Kpad, lda=384
__global__ void build_xk(const unsigned short* __restrict__ X0,
                         const unsigned short* __restrict__ X1,
                         const unsigned short* __restrict__ X2,
                         unsigned short* __restrict__ Xk, int F0, int Kpad, int ldx)
{
  const int idx = blockIdx.x * 256 + threadIdx.x;
  if (idx >= kM * Kpad) return;
  const int c = idx % Kpad;
  const int r = idx / Kpad;
  unsigned short v = 0;
  if (c < 3 * F0) {
    const int f = c / 3, m = c - 3 * f;
    const int n = r >> 5, b = r & 31;
    const unsigned short* src = (m == 0) ? X0 : (m == 1 ? X1 : X2);
    v = src[(size_t)n * ldx + f * kB + b];
  }
  Xk[(size_t)r * 384 + c] = v;
}

__global__ void rh_mul(const float* __restrict__ ru, const float* __restrict__ h,
                       float* __restrict__ rh)
{
  const int idx = blockIdx.x * 256 + threadIdx.x;   // kM*kU
  const int r = idx >> 6, j = idx & 63;
  rh[idx] = ru[(size_t)r * 128 + j] * h[idx];
}

__global__ void h_update(const float* __restrict__ ru, const float* __restrict__ cand,
                         float* __restrict__ h)
{
  const int idx = blockIdx.x * 256 + threadIdx.x;   // kM*kU
  const int r = idx >> 6, j = idx & 63;
  const float u = ru[(size_t)r * 128 + 64 + j];
  h[idx] = u * h[idx] + (1.f - u) * cand[idx];
}

// y = h_top @ projW + projb; writes d_out slice (B,N) and GO (N*B) for next step
__global__ void proj(const float* __restrict__ h, const float* __restrict__ pw,
                     const float* __restrict__ pb, float* __restrict__ yout,
                     float* __restrict__ go)
{
  const int r = blockIdx.x * 256 + threadIdx.x;
  if (r >= kM) return;
  float acc = pb[0];
#pragma unroll
  for (int j = 0; j < kU; ++j) acc += h[(size_t)r * kU + j] * pw[j];
  go[r] = acc;
  const int n = r >> 5, b = r & 31;
  yout[(size_t)b * kN + n] = acc;
}

__global__ void cvt_bf16(const float* __restrict__ s, unsigned short* __restrict__ d, int n)
{
  const int i = blockIdx.x * 256 + threadIdx.x;
  if (i < n) d[i] = f2bf(s[i]);
}

__global__ void cvt_w_pad(const float* __restrict__ W, unsigned short* __restrict__ o,
                          int Kraw, int Kpad, int nc)
{
  const int i = blockIdx.x * 256 + threadIdx.x;
  if (i >= Kpad * nc) return;
  const int k = i / nc, c = i % nc;
  o[i] = (k < Kraw) ? f2bf(W[(size_t)k * nc + c]) : (unsigned short)0;
}

// ---------------------------------------------------------------------------
// Host orchestration
// ---------------------------------------------------------------------------
extern "C" void kernel_launch(void* const* d_in, const int* in_sizes, int n_in,
                              void* d_out, int out_size, void* d_ws, size_t ws_size,
                              hipStream_t stream)
{
  (void)in_sizes; (void)n_in; (void)out_size; (void)ws_size;
  const float* inputs  = (const float*)d_in[0];
  const float* support = (const float*)d_in[1];
  struct CellP { const float *Wg, *bg, *Wc, *bc; };
  CellP cp[4];   // enc0, enc1, dec0, dec1
  for (int i = 0; i < 4; ++i) {
    cp[i].Wg = (const float*)d_in[2 + i * 4 + 0];
    cp[i].bg = (const float*)d_in[2 + i * 4 + 1];
    cp[i].Wc = (const float*)d_in[2 + i * 4 + 2];
    cp[i].bc = (const float*)d_in[2 + i * 4 + 3];
  }
  const float* projW = (const float*)d_in[18];
  const float* projB = (const float*)d_in[19];

  static const int F0s[4]   = {66, 128, 65, 128};    // Din + U
  static const int Dins[4]  = {2, 64, 1, 64};
  static const int Kpads[4] = {224, 384, 224, 384};  // 3*F0 -> mult of 32
  static const int Ldxs[4]  = {2176, 4096, 2176, 4096};  // F0*32 -> mult of 128

  char* p = (char*)d_ws;
  auto alloc = [&](size_t bytes) -> char* {
    char* r = p; p += (bytes + 255) & ~(size_t)255; return r;
  };
  unsigned short* Sb = (unsigned short*)alloc((size_t)kN * kN * 2);
  unsigned short *Wgb[4], *Wcb[4];
  for (int i = 0; i < 4; ++i) {
    Wgb[i] = (unsigned short*)alloc((size_t)Kpads[i] * 128 * 2);
    Wcb[i] = (unsigned short*)alloc((size_t)Kpads[i] * 64 * 2);
  }
  float* h0   = (float*)alloc((size_t)kM * kU * 4);
  float* h1   = (float*)alloc((size_t)kM * kU * 4);
  float* rh   = (float*)alloc((size_t)kM * kU * 4);
  float* ru   = (float*)alloc((size_t)kM * 128 * 4);
  float* cand = (float*)alloc((size_t)kM * kU * 4);
  float* go   = (float*)alloc((size_t)kM * 4);
  unsigned short* X0 = (unsigned short*)alloc((size_t)kN * 4096 * 2);
  unsigned short* X1 = (unsigned short*)alloc((size_t)kN * 4096 * 2);
  unsigned short* X2 = (unsigned short*)alloc((size_t)kN * 4096 * 2);
  unsigned short* Xk = (unsigned short*)alloc((size_t)kM * 384 * 2);

  // per-launch conversions + state init (graph-capturable)
  cvt_bf16<<<(kN * kN + 255) / 256, 256, 0, stream>>>(support, Sb, kN * kN);
  for (int i = 0; i < 4; ++i) {
    cvt_w_pad<<<(Kpads[i] * 128 + 255) / 256, 256, 0, stream>>>(cp[i].Wg, Wgb[i], 3 * F0s[i], Kpads[i], 128);
    cvt_w_pad<<<(Kpads[i] * 64 + 255) / 256, 256, 0, stream>>>(cp[i].Wc, Wcb[i], 3 * F0s[i], Kpads[i], 64);
  }
  (void)hipMemsetAsync(h0, 0, (size_t)kM * kU * 4, stream);
  (void)hipMemsetAsync(h1, 0, (size_t)kM * kU * 4, stream);
  (void)hipMemsetAsync(go, 0, (size_t)kM * 4, stream);

  auto run_gconv = [&](int mode, const float* xsrc, const float* hmix, int ci,
                       const unsigned short* Wb, const float* bias, int nc, int act,
                       float* outp) {
    const int F0 = F0s[ci], Din = Dins[ci], Kp = Kpads[ci], ldx = Ldxs[ci];
    pack_x0<<<(kN * ldx) / 256, 256, 0, stream>>>(mode, xsrc, hmix, X0, F0, Din, ldx);
    dim3 g(ldx / 128, kN / 128);
    gemm_diffuse<<<g, 256, 0, stream>>>(Sb, X0, nullptr, X1, ldx, 0);
    gemm_diffuse<<<g, 256, 0, stream>>>(Sb, X1, X0, X2, ldx, 1);
    build_xk<<<(kM * Kp) / 256, 256, 0, stream>>>(X0, X1, X2, Xk, F0, Kp, ldx);
    dim3 gw(nc / 64, kM / 128);
    gemm_weight<<<gw, 256, 0, stream>>>(Xk, Wb, bias, outp, Kp, nc, act);
  };

  auto run_cell = [&](int mode, const float* xsrc, float* h, int ci) {
    run_gconv(mode, xsrc, h, ci, Wgb[ci], cp[ci].bg, 128, 0, ru);       // gates (sigmoid)
    rh_mul<<<kM * kU / 256, 256, 0, stream>>>(ru, h, rh);               // r * h
    run_gconv(mode, xsrc, rh, ci, Wcb[ci], cp[ci].bc, 64, 1, cand);     // candidate (tanh)
    h_update<<<kM * kU / 256, 256, 0, stream>>>(ru, cand, h);           // h = u*h + (1-u)*c
  };

  // encoder
  for (int t = 0; t < kSeq; ++t) {
    run_cell(0, inputs + (size_t)t * kB * kN * 2, h0, 0);
    run_cell(2, h0, h1, 1);
  }
  // decoder
  float* out = (float*)d_out;
  for (int t = 0; t < kHor; ++t) {
    run_cell(1, go, h0, 2);
    run_cell(2, h0, h1, 3);
    proj<<<kM / 256, 256, 0, stream>>>(h1, projW, projB, out + (size_t)t * kB * kN, go);
  }
}
